// EncoderBlock_63247688400954
// MI455X (gfx1250) — matrix operand order repaired and checked by reading, not compiled, not run on hardware
//
#include <hip/hip_runtime.h>

// ---------------------------------------------------------------------------
// Types for CDNA5 WMMA (gfx1250, wave32)
// ---------------------------------------------------------------------------
typedef __attribute__((ext_vector_type(16))) __bf16 v16bf;
typedef __attribute__((ext_vector_type(8)))  float  v8f;

union FragB16 { v16bf v; unsigned int u[8]; };

__device__ __forceinline__ unsigned short f2bf(float f) {
    unsigned int u = __float_as_uint(f);
    u += 0x7fffu + ((u >> 16) & 1u);   // round-to-nearest-even
    return (unsigned short)(u >> 16);
}

__device__ __forceinline__ unsigned int pack2bf(float a, float b) {
    return (unsigned int)f2bf(a) | ((unsigned int)f2bf(b) << 16);
}

__device__ __forceinline__ unsigned short f2h_bits(float f) {
    union { _Float16 h; unsigned short u; } c; c.h = (_Float16)f; return c.u;
}
__device__ __forceinline__ float h2f_bits(unsigned short u) {
    union { unsigned short u; _Float16 h; } c; c.u = u; return (float)c.h;
}

// A-matrix (16x32 bf16) per-lane k base for VGPR v, lane-half hi (ISA 7.12.2)
__device__ __forceinline__ int a_k0(int v, int hi) {
    return (v < 4) ? (hi * 8 + 2 * v) : (16 + hi * 8 + 2 * (v - 4));
}

// LDS byte offset of a __shared__ pointer (flat LDS addr keeps offset in [31:0])
__device__ __forceinline__ unsigned lds_off(const void* p) {
    return (unsigned)(unsigned long long)(size_t)p;
}

// 32 bytes global -> LDS via CDNA5 async-to-LDS (2 ops, tracked by ASYNCcnt)
__device__ __forceinline__ void async_copy_32B(unsigned lds, const void* g) {
    unsigned long long ga = (unsigned long long)(size_t)g;
    asm volatile("global_load_async_to_lds_b128 %0, %1, off\n\t"
                 "global_load_async_to_lds_b128 %0, %1, off offset:16"
                 :: "v"(lds), "v"(ga) : "memory");
}
__device__ __forceinline__ void async_wait_le4() {
    asm volatile("s_wait_asynccnt 0x4" ::: "memory");
}
__device__ __forceinline__ void async_wait_0() {
    asm volatile("s_wait_asynccnt 0x0" ::: "memory");
}

// ---------------------------------------------------------------------------
// Weight transpose + fp32->bf16 convert: src[K][N] -> dst[N][K] (bf16)
// ---------------------------------------------------------------------------
__global__ __launch_bounds__(256) void transpose_convert_bf16(
    const float* __restrict__ src, unsigned short* __restrict__ dst,
    int K, int N) {
    __shared__ float t[32][33];
    const int tx = threadIdx.x & 31;
    const int ty = threadIdx.x >> 5;
    const int n0 = blockIdx.x * 32;
    const int k0 = blockIdx.y * 32;
#pragma unroll
    for (int i = 0; i < 4; ++i)
        t[ty + i * 8][tx] = src[(size_t)(k0 + ty + i * 8) * N + n0 + tx];
    __syncthreads();
#pragma unroll
    for (int i = 0; i < 4; ++i)
        dst[(size_t)(n0 + ty + i * 8) * K + k0 + tx] = f2bf(t[tx][ty + i * 8]);
}

// ---------------------------------------------------------------------------
// Custom LayerNorm: alpha*(x-mean)/(std+eps)+bias, std unbiased (ddof=1)
// ---------------------------------------------------------------------------
__global__ __launch_bounds__(256) void layernorm_bf16(
    const float* __restrict__ x, const float* __restrict__ alpha,
    const float* __restrict__ beta, unsigned short* __restrict__ out, int D) {
    const int row = blockIdx.x;
    const int tid = threadIdx.x;
    const float* xr = x + (size_t)row * D;

    float v[4];
    float s1 = 0.f, s2 = 0.f;
#pragma unroll
    for (int i = 0; i < 4; ++i) {
        float f = xr[tid + i * 256];
        v[i] = f; s1 += f; s2 += f * f;
    }
    __shared__ float r1[256], r2[256];
    r1[tid] = s1; r2[tid] = s2;
    __syncthreads();
    for (int off = 128; off > 0; off >>= 1) {
        if (tid < off) { r1[tid] += r1[tid + off]; r2[tid] += r2[tid + off]; }
        __syncthreads();
    }
    const float mean = r1[0] / (float)D;
    const float var  = (r2[0] - (float)D * mean * mean) / (float)(D - 1);
    const float stdv = sqrtf(fmaxf(var, 0.f));
    const float inv  = 1.0f / (stdv + 1e-6f);
    unsigned short* orow = out + (size_t)row * D;
#pragma unroll
    for (int i = 0; i < 4; ++i) {
        int c = tid + i * 256;
        orow[c] = f2bf(alpha[c] * (v[i] - mean) * inv + beta[c]);
    }
}

// ---------------------------------------------------------------------------
// WMMA bf16 GEMM with double-buffered async-to-LDS pipeline.
// out = A[M,K] @ Bt[N,K]^T + bias (+res) (opt relu)
// Block tile 128x128, K-step 32. 8 waves in 4(M) x 2(N); wave tile 32x64.
// OUT_MODE: 0 = fp32 row-major, 1 = bf16 row-major, 2 = bf16 transposed [N][M]
// ---------------------------------------------------------------------------
#define LDS_STRIDE 40   // ushort; 80 bytes: 16B aligned rows, 16 distinct banks

template <bool RELU, int OUT_MODE, bool HAS_RES>
__global__ __launch_bounds__(256) void gemm_bf16_wmma(
    const unsigned short* __restrict__ A,    // M x K bf16
    const unsigned short* __restrict__ Bt,   // N x K bf16 (pre-transposed)
    const float* __restrict__ bias,          // N
    const float* __restrict__ res,           // M x N fp32 (optional)
    void* __restrict__ out, int M, int N, int Kdim) {
    __shared__ unsigned short As[2][128 * LDS_STRIDE];
    __shared__ unsigned short Bs[2][128 * LDS_STRIDE];

    const int tid  = threadIdx.x;
    const int wave = tid >> 5;
    const int lane = tid & 31;
    const int l16  = lane & 15;
    const int hi   = lane >> 4;
    const int mtile = blockIdx.y * 128;
    const int ntile = blockIdx.x * 128;
    const int wm = wave >> 1;  // 4 M-slots of 32
    const int wn = wave & 1;   // 2 N-slots of 64

    v8f zero = {0.f, 0.f, 0.f, 0.f, 0.f, 0.f, 0.f, 0.f};
    v8f acc[2][4];
#pragma unroll
    for (int i = 0; i < 2; ++i)
#pragma unroll
        for (int j = 0; j < 4; ++j) acc[i][j] = zero;

    const int lr = tid >> 1;            // 0..127 tile row
    const int lk = (tid & 1) * 16;      // k-chunk 0/16

    const unsigned short* aG = A  + (size_t)(mtile + lr) * Kdim + lk;
    const unsigned short* bG = Bt + (size_t)(ntile + lr) * Kdim + lk;
    unsigned aL[2] = { lds_off(&As[0][lr * LDS_STRIDE + lk]),
                       lds_off(&As[1][lr * LDS_STRIDE + lk]) };
    unsigned bL[2] = { lds_off(&Bs[0][lr * LDS_STRIDE + lk]),
                       lds_off(&Bs[1][lr * LDS_STRIDE + lk]) };

    const int nk = Kdim >> 5;
    // prefetch tile 0
    async_copy_32B(aL[0], aG);
    async_copy_32B(bL[0], bG);

    for (int t = 0; t < nk; ++t) {
        const int cur = t & 1;
        if (t + 1 < nk) {
            async_copy_32B(aL[cur ^ 1], aG + (size_t)(t + 1) * 32);
            async_copy_32B(bL[cur ^ 1], bG + (size_t)(t + 1) * 32);
            async_wait_le4();     // tile t complete (in-order), t+1 in flight
        } else {
            async_wait_0();
        }
        __syncthreads();

        FragB16 af[2], bfr[4];
#pragma unroll
        for (int i = 0; i < 2; ++i) {
            const unsigned short* ap = &As[cur][(wm * 32 + i * 16 + l16) * LDS_STRIDE];
#pragma unroll
            for (int v = 0; v < 8; ++v)
                af[i].u[v] = *(const unsigned int*)(ap + a_k0(v, hi));
        }
#pragma unroll
        for (int j = 0; j < 4; ++j) {
            const unsigned short* bp = &Bs[cur][(wn * 64 + j * 16 + l16) * LDS_STRIDE];
#pragma unroll
            for (int v = 0; v < 8; ++v)
                bfr[j].u[v] = *(const unsigned int*)(bp + a_k0(v, hi));
        }
#pragma unroll
        for (int i = 0; i < 2; ++i)
#pragma unroll
            for (int j = 0; j < 4; ++j)
                acc[i][j] = __builtin_amdgcn_wmma_f32_16x16x32_bf16(
                    false, af[i].v, false, bfr[j].v, (short)0, acc[i][j], false, false);
        __syncthreads();
    }

    // Epilogue. C layout: VGPR r holds row (hi*8 + r), col = l16 (ISA 7.12.2)
#pragma unroll
    for (int i = 0; i < 2; ++i)
#pragma unroll
        for (int j = 0; j < 4; ++j) {
            const int row0 = mtile + wm * 32 + i * 16 + hi * 8;
            const int col  = ntile + wn * 64 + j * 16 + l16;
            const float bv = bias[col];
            if (OUT_MODE == 2) {
                unsigned int pk[4];
#pragma unroll
                for (int p = 0; p < 4; ++p)
                    pk[p] = pack2bf(acc[i][j][2 * p] + bv, acc[i][j][2 * p + 1] + bv);
                uint4 d; d.x = pk[0]; d.y = pk[1]; d.z = pk[2]; d.w = pk[3];
                *(uint4*)((unsigned short*)out + (size_t)col * M + row0) = d;
            } else {
#pragma unroll
                for (int r = 0; r < 8; ++r) {
                    const size_t idx = (size_t)(row0 + r) * N + col;
                    float val = acc[i][j][r] + bv;
                    if (RELU) val = fmaxf(val, 0.f);
                    if (HAS_RES) val += res[idx];
                    if (OUT_MODE == 1) ((unsigned short*)out)[idx] = f2bf(val);
                    else               ((float*)out)[idx] = val;
                }
            }
        }
}

// ---------------------------------------------------------------------------
// Flash attention with double-buffered async K/V tiles.
// Q/K bf16 [B*S,1024] (head h = cols h*64..); Vt bf16 [1024,B*S].
// Scores staged as fp16 in Ps, softmax overwrites in place with bf16 probs.
// ---------------------------------------------------------------------------
#define FA_STRIDE 72    // ushort; 144 bytes: 16B aligned, 16 distinct banks

__global__ __launch_bounds__(256) void flash_attn_bf16(
    const unsigned short* __restrict__ Q, const unsigned short* __restrict__ Km,
    const unsigned short* __restrict__ Vt, unsigned short* __restrict__ O,
    int ld, int Mtok, float scale) {
    __shared__ unsigned short Qs[64 * FA_STRIDE];
    __shared__ unsigned short Ks[2][64 * FA_STRIDE];   // [key][dk]
    __shared__ unsigned short Vs[2][64 * FA_STRIDE];   // [dk][key]
    __shared__ unsigned short Ps[64 * FA_STRIDE];      // f16 scores -> bf16 probs
    __shared__ float m_sh[64], l_sh[64], al_sh[64];

    const int tid  = threadIdx.x;
    const int wave = tid >> 5;
    const int lane = tid & 31;
    const int l16  = lane & 15;
    const int hi   = lane >> 4;

    const int bh = blockIdx.y;
    const size_t rowbase = (size_t)(bh >> 4) * 1024;
    const int colbase = (bh & 15) * 64;
    const int qrow0 = blockIdx.x * 64;

    const int om = wave >> 1;  // 4 row slots of 16
    const int on = wave & 1;   // 2 col slots of 32

    const int trow = tid >> 2;            // 0..63
    const int tcol = (tid & 3) * 16;      // 0/16/32/48

    // Load Q tile (64x64 bf16), synchronous
    {
        const uint4* src = (const uint4*)(Q + (rowbase + qrow0 + trow) * ld + colbase + tcol);
        uint4 d0 = src[0], d1 = src[1];
        uint4* dst = (uint4*)&Qs[trow * FA_STRIDE + tcol];
        dst[0] = d0; dst[1] = d1;
    }
    if (tid < 64) { m_sh[tid] = -1e30f; l_sh[tid] = 0.f; }

    const unsigned short* kG = Km + (rowbase + trow) * ld + colbase + tcol;
    const unsigned short* vG = Vt + (size_t)(colbase + trow) * Mtok + rowbase + tcol;
    unsigned kL[2] = { lds_off(&Ks[0][trow * FA_STRIDE + tcol]),
                       lds_off(&Ks[1][trow * FA_STRIDE + tcol]) };
    unsigned vL[2] = { lds_off(&Vs[0][trow * FA_STRIDE + tcol]),
                       lds_off(&Vs[1][trow * FA_STRIDE + tcol]) };

    v8f zero = {0.f,0.f,0.f,0.f,0.f,0.f,0.f,0.f};
    v8f oacc[2]; oacc[0] = zero; oacc[1] = zero;

    // prefetch key-tile 0
    async_copy_32B(kL[0], kG);
    async_copy_32B(vL[0], vG);
    __syncthreads();

    for (int j = 0; j < 16; ++j) {
        const int cur = j & 1;
        if (j + 1 < 16) {
            async_copy_32B(kL[cur ^ 1], kG + (size_t)(j + 1) * 64 * ld);
            async_copy_32B(vL[cur ^ 1], vG + (size_t)(j + 1) * 64);
            async_wait_le4();
        } else {
            async_wait_0();
        }
        __syncthreads();

        // S = scale * Q @ K^T ; stage as fp16 into Ps
        {
            v8f sacc[2]; sacc[0] = zero; sacc[1] = zero;
#pragma unroll
            for (int kk = 0; kk < 64; kk += 32) {
                FragB16 aq;
                const unsigned short* ap = &Qs[(om * 16 + l16) * FA_STRIDE + kk];
#pragma unroll
                for (int v = 0; v < 8; ++v)
                    aq.u[v] = *(const unsigned int*)(ap + a_k0(v, hi));
#pragma unroll
                for (int jj = 0; jj < 2; ++jj) {
                    FragB16 bk;
                    const unsigned short* bp = &Ks[cur][(on * 32 + jj * 16 + l16) * FA_STRIDE + kk];
#pragma unroll
                    for (int v = 0; v < 8; ++v)
                        bk.u[v] = *(const unsigned int*)(bp + a_k0(v, hi));
                    sacc[jj] = __builtin_amdgcn_wmma_f32_16x16x32_bf16(
                        false, aq.v, false, bk.v, (short)0, sacc[jj], false, false);
                }
            }
#pragma unroll
            for (int jj = 0; jj < 2; ++jj)
#pragma unroll
                for (int r = 0; r < 8; ++r)
                    Ps[(om * 16 + hi * 8 + r) * FA_STRIDE + on * 32 + jj * 16 + l16] =
                        f2h_bits(sacc[jj][r] * scale);
        }
        __syncthreads();

        // Online softmax per row (threads 0..63): f16 scores -> bf16 probs
        if (tid < 64) {
            unsigned int* prow = (unsigned int*)&Ps[tid * FA_STRIDE];
            const float mold = m_sh[tid];
            float mx = mold;
            for (int c2 = 0; c2 < 32; ++c2) {
                unsigned u = prow[c2];
                mx = fmaxf(mx, fmaxf(h2f_bits(u & 0xffffu), h2f_bits(u >> 16)));
            }
            float sum = 0.f;
            for (int c2 = 0; c2 < 32; ++c2) {
                unsigned u = prow[c2];
                float p0 = __expf(h2f_bits(u & 0xffffu) - mx);
                float p1 = __expf(h2f_bits(u >> 16)     - mx);
                sum += p0 + p1;
                prow[c2] = pack2bf(p0, p1);
            }
            float a = __expf(mold - mx);
            l_sh[tid] = l_sh[tid] * a + sum;
            m_sh[tid] = mx;
            al_sh[tid] = a;
        }
        __syncthreads();

        // Rescale O, then O += P @ V
        {
            float ar[8];
#pragma unroll
            for (int r = 0; r < 8; ++r) ar[r] = al_sh[om * 16 + hi * 8 + r];
#pragma unroll
            for (int jj = 0; jj < 2; ++jj)
#pragma unroll
                for (int r = 0; r < 8; ++r) oacc[jj][r] *= ar[r];
#pragma unroll
            for (int kk = 0; kk < 64; kk += 32) {
                FragB16 ap2;
                const unsigned short* pp = &Ps[(om * 16 + l16) * FA_STRIDE + kk];
#pragma unroll
                for (int v = 0; v < 8; ++v)
                    ap2.u[v] = *(const unsigned int*)(pp + a_k0(v, hi));
#pragma unroll
                for (int jj = 0; jj < 2; ++jj) {
                    FragB16 bv;
                    const unsigned short* bp = &Vs[cur][(on * 32 + jj * 16 + l16) * FA_STRIDE + kk];
#pragma unroll
                    for (int v = 0; v < 8; ++v)
                        bv.u[v] = *(const unsigned int*)(bp + a_k0(v, hi));
                    oacc[jj] = __builtin_amdgcn_wmma_f32_16x16x32_bf16(
                        false, ap2.v, false, bv.v, (short)0, oacc[jj], false, false);
                }
            }
        }
        __syncthreads();
    }

    // Finalize: O /= l, write bf16
#pragma unroll
    for (int jj = 0; jj < 2; ++jj)
#pragma unroll
        for (int r = 0; r < 8; ++r) {
            const int row = om * 16 + hi * 8 + r;
            const float inv = 1.0f / l_sh[row];
            O[(rowbase + qrow0 + row) * ld + colbase + on * 32 + jj * 16 + l16] =
                f2bf(oacc[jj][r] * inv);
        }
}

// ---------------------------------------------------------------------------
// Host-side orchestration
// ---------------------------------------------------------------------------
extern "C" void kernel_launch(void* const* d_in, const int* in_sizes, int n_in,
                              void* d_out, int out_size, void* d_ws, size_t ws_size,
                              hipStream_t stream) {
    (void)in_sizes; (void)n_in; (void)out_size; (void)ws_size;
    const int D = 1024, F = 4096, Mrows = 8 * 1024;  // 8192 tokens

    const float* x    = (const float*)d_in[0];
    const float* wq   = (const float*)d_in[1];
    const float* bq   = (const float*)d_in[2];
    const float* wk   = (const float*)d_in[3];
    const float* bk   = (const float*)d_in[4];
    const float* wv   = (const float*)d_in[5];
    const float* bv   = (const float*)d_in[6];
    const float* wo   = (const float*)d_in[7];
    const float* bo   = (const float*)d_in[8];
    const float* w1   = (const float*)d_in[9];
    const float* b1   = (const float*)d_in[10];
    const float* w2   = (const float*)d_in[11];
    const float* b2   = (const float*)d_in[12];
    const float* ln1a = (const float*)d_in[13];
    const float* ln1b = (const float*)d_in[14];
    const float* ln2a = (const float*)d_in[15];
    const float* ln2b = (const float*)d_in[16];

    char* ws = (char*)d_ws;
    unsigned short* WQ  = (unsigned short*)(ws + 0);          // weights bf16,
    unsigned short* WK  = (unsigned short*)(ws + 2097152);    // pre-transposed [N][K]
    unsigned short* WV  = (unsigned short*)(ws + 4194304);
    unsigned short* WO  = (unsigned short*)(ws + 6291456);
    unsigned short* W1  = (unsigned short*)(ws + 8388608);
    unsigned short* W2  = (unsigned short*)(ws + 16777216);
    unsigned short* Z   = (unsigned short*)(ws + 25165824);   // 8192x1024 bf16
    float*          Hf  = (float*)         (ws + 41943040);   // 8192x1024 f32
    unsigned short* Qb  = (unsigned short*)(ws + 75497472);
    unsigned short* Kb  = (unsigned short*)(ws + 92274688);
    unsigned short* Vtb = (unsigned short*)(ws + 109051904);  // V^T [1024][8192]
    unsigned short* ATT = (unsigned short*)(ws + 125829120);
    unsigned short* FF1 = (unsigned short*)(ws + 75497472);   // reuses Q/K/V/ATT

    // 1. weights -> bf16, transposed to [N][K]
    dim3 gtw(D / 32, D / 32);
    transpose_convert_bf16<<<gtw, 256, 0, stream>>>(wq, WQ, D, D);
    transpose_convert_bf16<<<gtw, 256, 0, stream>>>(wk, WK, D, D);
    transpose_convert_bf16<<<gtw, 256, 0, stream>>>(wv, WV, D, D);
    transpose_convert_bf16<<<gtw, 256, 0, stream>>>(wo, WO, D, D);
    transpose_convert_bf16<<<dim3(F / 32, D / 32), 256, 0, stream>>>(w1, W1, D, F);
    transpose_convert_bf16<<<dim3(D / 32, F / 32), 256, 0, stream>>>(w2, W2, F, D);

    // 2. LayerNorm1 -> bf16
    layernorm_bf16<<<Mrows, 256, 0, stream>>>(x, ln1a, ln1b, Z, D);

    // 3. Q/K/V projections (V emitted transposed for flash's [dk][key] tiles)
    dim3 g1(D / 128, Mrows / 128);
    gemm_bf16_wmma<false, 1, false><<<g1, 256, 0, stream>>>(Z, WQ, bq, nullptr, Qb, Mrows, D, D);
    gemm_bf16_wmma<false, 1, false><<<g1, 256, 0, stream>>>(Z, WK, bk, nullptr, Kb, Mrows, D, D);
    gemm_bf16_wmma<false, 2, false><<<g1, 256, 0, stream>>>(Z, WV, bv, nullptr, Vtb, Mrows, D, D);

    // 4. Flash attention (scale = 1/sqrt(64))
    dim3 ga(1024 / 64, 8 * 16);
    flash_attn_bf16<<<ga, 256, 0, stream>>>(Qb, Kb, Vtb, ATT, D, Mrows, 0.125f);

    // 5. Output projection + residual(x) -> h (fp32)
    gemm_bf16_wmma<false, 0, true><<<g1, 256, 0, stream>>>(ATT, WO, bo, x, Hf, Mrows, D, D);

    // 6. LayerNorm2 -> bf16
    layernorm_bf16<<<Mrows, 256, 0, stream>>>(Hf, ln2a, ln2b, Z, D);

    // 7. FFN up + ReLU -> bf16
    dim3 g2(F / 128, Mrows / 128);
    gemm_bf16_wmma<true, 1, false><<<g2, 256, 0, stream>>>(Z, W1, b1, nullptr, FF1, Mrows, F, D);

    // 8. FFN down + bias + residual(h) -> d_out (fp32)
    gemm_bf16_wmma<false, 0, true><<<g1, 256, 0, stream>>>(FF1, W2, b2, Hf, (float*)d_out, Mrows, D, F);
}
